// AttentionSequencePoolingLayer_19490561589427
// MI455X (gfx1250) — matrix operand order repaired and checked by reading, not compiled, not run on hardware
//
#include <hip/hip_runtime.h>
#include <hip/hip_bf16.h>
#include <math.h>

// ---- shapes ----
#define BB    2048
#define LL    200
#define DD    128
#define H1    128
#define H2    32
#define RTOT  (BB*LL)      // 409600 rows, divisible by 128
#define EPSV  1e-8f

typedef __attribute__((ext_vector_type(16))) __bf16 v16bf;
typedef __attribute__((ext_vector_type(8)))  __bf16 v8bf;
typedef __attribute__((ext_vector_type(8)))  float  v8f;
typedef __attribute__((ext_vector_type(4)))  float  v4f;

static __device__ __forceinline__ __bf16 f2bf(float f) { return (__bf16)f; }
static __device__ __forceinline__ float  bf2f(__bf16 b){ return (float)b; }
static __device__ __forceinline__ float  sigm(float x) { return 1.0f/(1.0f + __expf(-x)); }

// ---------------------------------------------------------------------------
// K1: fold & transpose weights to bf16; zero the global stat accumulators.
// wcat[n][k] (n=H1 channel, k=0..255): k<128 -> W1[128+k][n]-W1[256+k][n] (keys)
//                                      k>=128 -> W1[384+(k-128)][n]      (q*k)
// w2t[n][k]  (n=H2 channel, k=0..127): W2[k][n]
// ---------------------------------------------------------------------------
__global__ void k_prep(const float* __restrict__ W1, const float* __restrict__ W2,
                       __bf16* __restrict__ wcat, __bf16* __restrict__ w2t,
                       float* sum1, float* ss1, float* sum2, float* ss2) {
  int t = blockIdx.x*blockDim.x + threadIdx.x;
  if (t < 128*256) {
    int n = t >> 8, k = t & 255;
    float v;
    if (k < 128) v = W1[(128+k)*H1 + n] - W1[(256+k)*H1 + n];
    else         v = W1[(384 + (k-128))*H1 + n];
    wcat[n*256 + k] = f2bf(v);
  }
  int t2 = t - 128*256;
  if (t2 >= 0 && t2 < 32*128) {
    int n = t2 >> 7, k = t2 & 127;
    w2t[n*128 + k] = f2bf(W2[k*H2 + n]);
  }
  if (t < H1) { sum1[t] = 0.f; ss1[t] = 0.f; }
  if (t < H2) { sum2[t] = 0.f; ss2[t] = 0.f; }
}

// ---------------------------------------------------------------------------
// K2: per-batch q contribution: qbase[b][h] = b1[h] + sum_d q[b][d]*(W1a+W1c)[d][h]
// ---------------------------------------------------------------------------
__global__ void k_qbase(const float* __restrict__ q, const float* __restrict__ W1,
                        const float* __restrict__ b1, float* __restrict__ qbase) {
  __shared__ float sq[DD];
  int b = blockIdx.x, h = threadIdx.x;
  sq[h] = q[(long)b*DD + h];
  __syncthreads();
  float acc = b1[h];
  #pragma unroll 4
  for (int d = 0; d < DD; ++d)
    acc += sq[d] * (W1[d*H1 + h] + W1[(256+d)*H1 + h]);
  qbase[(long)b*H1 + h] = acc;
}

// ---------------------------------------------------------------------------
// K3: GEMM1 with WMMA bf16: z1 = [k | q*k] @ [Wk;Wp] + qbase, accumulate stats.
// 8 waves/WG, each wave owns a 16-row M-tile; W staged in LDS in two K-halves.
// ---------------------------------------------------------------------------
__global__ void __launch_bounds__(256) k_gemm1(
    const float* __restrict__ query, const float* __restrict__ keys,
    const __bf16* __restrict__ wcat, const float* __restrict__ qbase,
    __bf16* __restrict__ z1, float* __restrict__ gsum1, float* __restrict__ gss1) {
  __shared__ __attribute__((aligned(16))) __bf16 sW[128*128];  // one K-half: [n][k] 32KB
  __shared__ float sSum[H1], sSS[H1];
  const int tid  = threadIdx.x;
  const int lane = tid & 31;
  const int wave = tid >> 5;
  if (tid < H1) { sSum[tid] = 0.f; sSS[tid] = 0.f; }

  const int  mrow = lane & 15;
  const bool hi   = lane >= 16;
  const int  ko   = hi ? 16 : 0;
  const long row0 = (long)(blockIdx.x*8 + wave) * 16;
  const long r    = row0 + mrow;                       // flat (b*L+l) row
  const int  bidx = (int)((unsigned)r / 200u);
  const float* krow = keys + r*DD;                     // keys row == flat row
  const float* qrow = query + (long)bidx*DD;

  v8f acc[8] = {};

  for (int half = 0; half < 2; ++half) {
    __syncthreads();
    // stage 128x128 bf16 half of Wcat^T into LDS (16B per thread per iter)
    for (int i = tid; i < 2048; i += 256) {
      int e = i*8, n = e >> 7, k = e & 127;
      *(uint4*)(&sW[e]) = *(const uint4*)(&wcat[n*256 + half*128 + k]);
    }
    __syncthreads();

    #pragma unroll
    for (int kk = 0; kk < 4; ++kk) {
      // Build A operand (16x32 bf16): lane row = lane%16; elems 0..15 = K+{0..15},
      // hi lanes shift K by 16.
      v16bf a;
      #pragma unroll
      for (int seg = 0; seg < 2; ++seg) {
        const int d = kk*32 + ko + seg*8;              // within this K-half
        v4f x0 = *(const v4f*)(krow + d);
        v4f x1 = *(const v4f*)(krow + d + 4);
        if (half) {                                    // q*k features
          v4f q0 = *(const v4f*)(qrow + d);
          v4f q1 = *(const v4f*)(qrow + d + 4);
          x0 *= q0; x1 *= q1;
        }
        a[seg*8+0]=f2bf(x0.x); a[seg*8+1]=f2bf(x0.y); a[seg*8+2]=f2bf(x0.z); a[seg*8+3]=f2bf(x0.w);
        a[seg*8+4]=f2bf(x1.x); a[seg*8+5]=f2bf(x1.y); a[seg*8+6]=f2bf(x1.z); a[seg*8+7]=f2bf(x1.w);
      }
      const int koff = kk*32 + (hi ? 16 : 0);          // B: lane holds 16-deep K col seg
      #pragma unroll
      for (int n = 0; n < 8; ++n) {
        v16bf bmat = *(const v16bf*)(&sW[(n*16 + mrow)*128 + koff]);
        acc[n] = __builtin_amdgcn_wmma_f32_16x16x32_bf16(
            false, a, false, bmat, (short)0, acc[n], false, false);
      }
    }
  }

  // epilogue: + qbase, store bf16 z1, per-channel partial stats
  const int  b0   = (int)((unsigned)row0 / 200u);
  const int  b1i  = (int)((unsigned)(row0 + 15) / 200u);
  const long lim  = (long)(b0 + 1) * 200;
  #pragma unroll
  for (int n = 0; n < 8; ++n) {
    const int ch = n*16 + mrow;                        // D layout: lane = N
    const float qv0 = qbase[(long)b0*H1 + ch];
    const float qv1 = qbase[(long)b1i*H1 + ch];
    float s = 0.f, s2 = 0.f;
    #pragma unroll
    for (int j = 0; j < 8; ++j) {                      // D layout: VGPR j = row M (+8 if hi)
      long rr = row0 + (hi ? 8 : 0) + j;
      float z = acc[n][j] + (rr < lim ? qv0 : qv1);
      z1[rr*H1 + ch] = f2bf(z);
      s += z; s2 += z*z;
    }
    atomicAdd(&sSum[ch], s);
    atomicAdd(&sSS[ch], s2);
  }
  __syncthreads();
  if (tid < H1) { atomicAdd(&gsum1[tid], sSum[tid]); atomicAdd(&gss1[tid], sSS[tid]); }
}

// ---------------------------------------------------------------------------
// K4: finalize batch stats -> mean, rsqrt(var+eps)
// ---------------------------------------------------------------------------
__global__ void k_fin(const float* __restrict__ sum, const float* __restrict__ ss,
                      float* __restrict__ mean, float* __restrict__ scale, int nch) {
  int c = threadIdx.x;
  if (c < nch) {
    float m = sum[c] * (1.0f / (float)RTOT);
    float v = ss[c] * (1.0f / (float)RTOT) - m*m;
    mean[c]  = m;
    scale[c] = rsqrtf(v + EPSV);
  }
}

// ---------------------------------------------------------------------------
// K5: dice(z1) fused into A-load, GEMM2 via WMMA bf16, z2 fp32 + stats.
// ---------------------------------------------------------------------------
__global__ void __launch_bounds__(256) k_gemm2(
    const __bf16* __restrict__ z1, const __bf16* __restrict__ w2t,
    const float* __restrict__ mean1, const float* __restrict__ scale1,
    const float* __restrict__ alpha1, const float* __restrict__ b2,
    float* __restrict__ z2, float* __restrict__ gsum2, float* __restrict__ gss2) {
  __shared__ __attribute__((aligned(16))) __bf16 sW[32*128];   // W2^T bf16, 8KB
  __shared__ float sM[H1], sSc[H1], sA[H1];
  __shared__ float sSum[H2], sSS[H2];
  const int tid = threadIdx.x, lane = tid & 31, wave = tid >> 5;
  if (tid < H1) { sM[tid] = mean1[tid]; sSc[tid] = scale1[tid]; sA[tid] = alpha1[tid]; }
  if (tid < H2) { sSum[tid] = 0.f; sSS[tid] = 0.f; }
  for (int i = tid*8; i < 32*128; i += 256*8)
    *(uint4*)(&sW[i]) = *(const uint4*)(&w2t[i]);
  __syncthreads();

  const int  mrow = lane & 15;
  const bool hi   = lane >= 16;
  const int  ko   = hi ? 16 : 0;
  const long row0 = (long)(blockIdx.x*8 + wave) * 16;
  const long r    = row0 + mrow;
  const __bf16* zrow = z1 + r*H1;

  v8f acc[2] = {};
  #pragma unroll
  for (int kk = 0; kk < 4; ++kk) {
    v16bf a;
    #pragma unroll
    for (int seg = 0; seg < 2; ++seg) {
      const int d = kk*32 + ko + seg*8;
      v8bf zv = *(const v8bf*)(zrow + d);
      #pragma unroll
      for (int i = 0; i < 8; ++i) {
        const int c = d + i;
        float z = bf2f(zv[i]);
        float p = sigm((z - sM[c]) * sSc[c]);
        float h = z * (p + sA[c] * (1.f - p));
        a[seg*8 + i] = f2bf(h);
      }
    }
    const int koff = kk*32 + (hi ? 16 : 0);
    #pragma unroll
    for (int n = 0; n < 2; ++n) {
      v16bf bmat = *(const v16bf*)(&sW[(n*16 + mrow)*128 + koff]);
      acc[n] = __builtin_amdgcn_wmma_f32_16x16x32_bf16(
          false, a, false, bmat, (short)0, acc[n], false, false);
    }
  }

  #pragma unroll
  for (int n = 0; n < 2; ++n) {
    const int ch = n*16 + mrow;
    const float bias = b2[ch];
    float s = 0.f, s2 = 0.f;
    #pragma unroll
    for (int j = 0; j < 8; ++j) {
      long rr = row0 + (hi ? 8 : 0) + j;
      float z = acc[n][j] + bias;
      z2[rr*H2 + ch] = z;
      s += z; s2 += z*z;
    }
    atomicAdd(&sSum[ch], s);
    atomicAdd(&sSS[ch], s2);
  }
  __syncthreads();
  if (tid < H2) { atomicAdd(&gsum2[tid], sSum[tid]); atomicAdd(&gss2[tid], sSS[tid]); }
}

// ---------------------------------------------------------------------------
// K7: dice(z2) -> score, mask by keys_len, weighted-sum pooling over L.
// One workgroup per batch row b; keys row reads fully coalesced.
// ---------------------------------------------------------------------------
__global__ void __launch_bounds__(256) k_pool(
    const float* __restrict__ z2, const float* __restrict__ keys,
    const int* __restrict__ keys_len,
    const float* __restrict__ mean2, const float* __restrict__ scale2,
    const float* __restrict__ alpha2,
    const float* __restrict__ Wd, const float* __restrict__ bd,
    float* __restrict__ out) {
  __shared__ float sS[LL];
  __shared__ float sPart[256];
  __shared__ float sM[H2], sSc[H2], sA[H2], sWd[H2];
  const int tid = threadIdx.x, b = blockIdx.x;
  if (tid < H2) { sM[tid]=mean2[tid]; sSc[tid]=scale2[tid]; sA[tid]=alpha2[tid]; sWd[tid]=Wd[tid]; }
  __syncthreads();
  if (tid < LL) {
    const long row = (long)b*LL + tid;
    float acc = bd[0];
    #pragma unroll
    for (int c = 0; c < H2; ++c) {
      float z = z2[row*H2 + c];
      float p = sigm((z - sM[c]) * sSc[c]);
      float h = z * (p + sA[c] * (1.f - p));
      acc += h * sWd[c];
    }
    sS[tid] = (tid < keys_len[b]) ? acc : 0.f;
  }
  __syncthreads();
  const int d = tid & 127, hsel = tid >> 7;
  const float* kb = keys + ((long)b*LL + hsel*100) * DD;
  float acc = 0.f;
  for (int l = 0; l < 100; ++l)
    acc += sS[hsel*100 + l] * kb[(long)l*DD + d];
  sPart[tid] = acc;
  __syncthreads();
  if (tid < DD) out[(long)b*DD + tid] = sPart[tid] + sPart[128 + tid];
}

// ---------------------------------------------------------------------------
extern "C" void kernel_launch(void* const* d_in, const int* in_sizes, int n_in,
                              void* d_out, int out_size, void* d_ws, size_t ws_size,
                              hipStream_t stream) {
  const float* query    = (const float*)d_in[0];
  const float* keys     = (const float*)d_in[1];
  const int*   keys_len = (const int*)  d_in[2];
  const float* W1       = (const float*)d_in[3];
  const float* b1       = (const float*)d_in[4];
  const float* alpha1   = (const float*)d_in[5];
  const float* W2       = (const float*)d_in[6];
  const float* b2       = (const float*)d_in[7];
  const float* alpha2   = (const float*)d_in[8];
  const float* Wd       = (const float*)d_in[9];
  const float* bd       = (const float*)d_in[10];
  float* out = (float*)d_out;

  char* ws = (char*)d_ws;
  __bf16* wcat   = (__bf16*)(ws + 0);            // 65536
  __bf16* w2t    = (__bf16*)(ws + 65536);        // 8192
  float*  qbase  = (float*) (ws + 73728);        // 1048576
  float*  sum1   = (float*) (ws + 1122304);
  float*  ss1    = (float*) (ws + 1122816);
  float*  mean1  = (float*) (ws + 1123328);
  float*  scale1 = (float*) (ws + 1123840);
  float*  sum2   = (float*) (ws + 1124352);
  float*  ss2    = (float*) (ws + 1124608);
  float*  mean2  = (float*) (ws + 1124864);
  float*  scale2 = (float*) (ws + 1125120);
  __bf16* z1     = (__bf16*)(ws + 1125376);                       // 104857600 (bf16)
  float*  z2     = (float*) (ws + 1125376 + 104857600ull);        // 52428800

  k_prep <<<dim3(145),  dim3(256), 0, stream>>>(W1, W2, wcat, w2t, sum1, ss1, sum2, ss2);
  k_qbase<<<dim3(BB),   dim3(128), 0, stream>>>(query, W1, b1, qbase);
  k_gemm1<<<dim3(RTOT/128), dim3(256), 0, stream>>>(query, keys, wcat, qbase, z1, sum1, ss1);
  k_fin  <<<dim3(1),    dim3(128), 0, stream>>>(sum1, ss1, mean1, scale1, H1);
  k_gemm2<<<dim3(RTOT/128), dim3(256), 0, stream>>>(z1, w2t, mean1, scale1, alpha1, b2, z2, sum2, ss2);
  k_fin  <<<dim3(1),    dim3(32),  0, stream>>>(sum2, ss2, mean2, scale2, H2);
  k_pool <<<dim3(BB),   dim3(256), 0, stream>>>(z2, keys, keys_len, mean2, scale2, alpha2, Wd, bd, out);
}
